// FPNDcnLconv3Dcn_67018669687099
// MI455X (gfx1250) — matrix operand order repaired and checked by reading, not compiled, not run on hardware
//
#include <hip/hip_runtime.h>
#include <hip/hip_bf16.h>

typedef __attribute__((ext_vector_type(16))) _Float16 v16h;
typedef __attribute__((ext_vector_type(8)))  float    v8f;
typedef int b128_t __attribute__((vector_size(16)));   // matches async-LDS builtin pointee

#if __has_builtin(__builtin_amdgcn_global_load_async_to_lds_b128)
#define ASYNC_LDS 1
#else
#define ASYNC_LDS 0
#endif

// ---------------------------------------------------------------------------
// Pack f32 weight matrix [M, Kdim] into WMMA A-fragment order (f16).
// A 16x32 f16 fragment, contiguous k order used here (same in B): lane = (m&15) + 16*h, h = bit4 of (k&31);
// j = k&15. Each lane's 16 halves are contiguous (32B).
// ---------------------------------------------------------------------------
__global__ void pack_A_f16(const float* __restrict__ w, unsigned short* __restrict__ ap_,
                           int M, int Kdim) {
  _Float16* __restrict__ ap = (_Float16*)ap_;
  long idx = (long)blockIdx.x * blockDim.x + threadIdx.x;
  long total = (long)M * Kdim;
  if (idx >= total) return;
  int m = (int)(idx / Kdim), k = (int)(idx % Kdim);
  int mt = m >> 4, mr = m & 15;
  int kt = k >> 5, kr = k & 31;
  int h  = kr >> 4;
  int j  = kr & 15;
  int lane = mr + (h << 4);
  int Ktiles = Kdim >> 5;
  ap[(((size_t)mt * Ktiles + kt) * 32 + lane) * 16 + j] = (_Float16)w[idx];
}

// ---------------------------------------------------------------------------
// Offset-prediction conv: plain 3x3, stride 1, pad 1, 18 output channels, bias.
// ---------------------------------------------------------------------------
__global__ void offset_conv(const float* __restrict__ x, const float* __restrict__ ow,
                            const float* __restrict__ ob, float* __restrict__ off,
                            int Bn, int Cin, int H, int W) {
  long idx = (long)blockIdx.x * blockDim.x + threadIdx.x;
  long total = (long)Bn * 18 * H * W;
  if (idx >= total) return;
  int xw = (int)(idx % W); long t = idx / W;
  int yh = (int)(t % H);   t /= H;
  int oc = (int)(t % 18);  int b = (int)(t / 18);
  float acc = ob[oc];
  const float* xb = x + (size_t)b * Cin * H * W;
  const float* wr = ow + (size_t)oc * Cin * 9;
  for (int c = 0; c < Cin; ++c) {
    const float* xc = xb + (size_t)c * H * W;
    const float* wc = wr + c * 9;
#pragma unroll
    for (int ky = 0; ky < 3; ++ky) {
      int yy = yh + ky - 1;
      if (yy < 0 || yy >= H) continue;
#pragma unroll
      for (int kx = 0; kx < 3; ++kx) {
        int xx = xw + kx - 1;
        if (xx < 0 || xx >= W) continue;
        acc += xc[yy * W + xx] * wc[ky * 3 + kx];
      }
    }
  }
  off[idx] = acc;
}

// ---------------------------------------------------------------------------
// Deformable im2col -> WMMA B-fragment order (f16). One lane's 16 contiguous
// halves = 16 consecutive K at fixed n, so each thread computes 16 consecutive
// kk and emits one contiguous 32B v16h store (512B per 16 threads).
// ---------------------------------------------------------------------------
__global__ void sample_pack_B(const float* __restrict__ x, const float* __restrict__ off,
                              unsigned short* __restrict__ bp_, int Cin, int H, int W) {
  _Float16* __restrict__ bp = (_Float16*)bp_;
  int HW = H * W;
  long idx = (long)blockIdx.x * blockDim.x + threadIdx.x;
  long total = ((long)Cin * 9 * HW) >> 4;
  if (idx >= total) return;
  int n    = (int)(idx % HW);
  int kc16 = (int)(idx / HW);
  int y = n / W, xw = n % W;

  _Float16 vals[16];
  int kkBase = kc16 << 4;
#pragma unroll
  for (int j = 0; j < 16; ++j) {
    int kk = kkBase + j;
    int c = kk / 9, kp = kk % 9;
    int ky = kp / 3, kx = kp % 3;
    float dy = off[(kp * 2 + 0) * HW + n];
    float dx = off[(kp * 2 + 1) * HW + n];
    float py = dy + (float)(ky + y - 1);
    float px = dx + (float)(kx + xw - 1);
    float y0f = floorf(py), x0f = floorf(px);
    float ly = py - y0f, lx = px - x0f;
    int y0 = (int)y0f, x0 = (int)x0f;
    const float* xc = x + (size_t)c * HW;
    float v = 0.f;
#pragma unroll
    for (int di = 0; di < 2; ++di) {
#pragma unroll
      for (int dj = 0; dj < 2; ++dj) {
        int yy = y0 + di, xx = x0 + dj;
        float wt = (di ? ly : 1.f - ly) * (dj ? lx : 1.f - lx);
        bool ok = (yy >= 0) && (yy < H) && (xx >= 0) && (xx < W);
        int ycl = yy < 0 ? 0 : (yy > H - 1 ? H - 1 : yy);
        int xcl = xx < 0 ? 0 : (xx > W - 1 ? W - 1 : xx);
        v += ok ? xc[ycl * W + xcl] * wt : 0.f;
      }
    }
    vals[j] = (_Float16)v;
  }
  int kt = kc16 >> 1;
  int lane = (n & 15) | ((kc16 & 1) << 4);
  int nt = n >> 4;
  int Ntiles = HW >> 4;
  *(v16h*)(bp + (((size_t)kt * Ntiles + nt) * 32 + lane) * 16) = *(const v16h*)vals;
}

#define KCHUNK 16                 // K-tiles staged per LDS round
#define APITCH 24                 // halves per lane-chunk in LDS (16 data + 8 pad)

// ---------------------------------------------------------------------------
// FULL GEMM (requires Ntiles % 32 == 0): block = 8 waves, one M-tile per
// block (blockIdx.y), 4 N-tiles per wave, no guards -> no exec manipulation.
// A staged in LDS via async-to-LDS; K loop ping-pongs between two register
// sets (kcnt is always even: every Ktiles is a multiple of 8) so the
// software pipeline needs no rotation copies.
// ---------------------------------------------------------------------------
__global__ void __launch_bounds__(256)
wmma_gemm_f16_full(const unsigned short* __restrict__ ap_, const unsigned short* __restrict__ bp_,
                   float* __restrict__ out, int Ntiles, int Ktiles, int N) {
  __shared__ _Float16 As[KCHUNK * 32 * APITCH];   // 24KB
  const _Float16* __restrict__ ap = (const _Float16*)ap_;
  const _Float16* __restrict__ bp = (const _Float16*)bp_;
  int waveId = __builtin_amdgcn_readfirstlane(threadIdx.x >> 5);  // force SGPR
  int lane   = threadIdx.x & 31;
  int mt     = blockIdx.y;
  int ntBase = (blockIdx.x * 8 + waveId) * 4;

  const _Float16* aBase  = ap + (size_t)mt * Ktiles * 512;
  const _Float16* AsLane = As + lane * APITCH;
  v8f c0 = {}, c1 = {}, c2 = {}, c3 = {};

  for (int kc = 0; kc < Ktiles; kc += KCHUNK) {
    int kcnt = Ktiles - kc;
    if (kcnt > KCHUNK) kcnt = KCHUNK;                // always even (8 or 16)
    __syncthreads();
    // ---- cooperative async stage of A chunk ----
    {
      int nvec = kcnt * 64;                          // 16B vectors
      for (int t = threadIdx.x; t < nvec; t += 256) {
        int chunk = t >> 1, half = t & 1;
        int kkL = chunk >> 5, sl = chunk & 31;
        const b128_t* s = (const b128_t*)(aBase + ((size_t)kc * 32 + chunk) * 16) + half;
        _Float16* d = As + kkL * (32 * APITCH) + sl * APITCH + half * 8;
#if ASYNC_LDS
        __builtin_amdgcn_global_load_async_to_lds_b128(
            (__attribute__((address_space(1))) b128_t*)s,
            (__attribute__((address_space(3))) b128_t*)d, 0, 0);
#else
        *(b128_t*)d = *s;
#endif
      }
    }
#if ASYNC_LDS
#if __has_builtin(__builtin_amdgcn_s_wait_asynccnt)
    __builtin_amdgcn_s_wait_asynccnt(0);
#else
    asm volatile("s_wait_asynccnt 0x0" ::: "memory");
#endif
#endif
    __syncthreads();

    if (kc + kcnt < Ktiles)
      __builtin_prefetch(aBase + (((size_t)kc + kcnt) * 32 + lane) * 16, 0, 3);

    // ---- ping-pong software-pipelined K loop (kcnt even) ----
    const _Float16* bT = bp + (((size_t)kc * Ntiles + ntBase) * 32 + lane) * 16;
    v16h a0  = *(const v16h*)(AsLane);
    v16h b00 = *(const v16h*)(bT + 0 * 512);
    v16h b01 = *(const v16h*)(bT + 1 * 512);
    v16h b02 = *(const v16h*)(bT + 2 * 512);
    v16h b03 = *(const v16h*)(bT + 3 * 512);
    for (int kk = 0; kk < kcnt; kk += 2) {
      // load odd set (kk+1) while computing even set (kk)
      const _Float16* bT1 =
          bp + (((size_t)(kc + kk + 1) * Ntiles + ntBase) * 32 + lane) * 16;
      v16h a1  = *(const v16h*)(AsLane + (kk + 1) * (32 * APITCH));
      v16h b10 = *(const v16h*)(bT1 + 0 * 512);
      v16h b11 = *(const v16h*)(bT1 + 1 * 512);
      v16h b12 = *(const v16h*)(bT1 + 2 * 512);
      v16h b13 = *(const v16h*)(bT1 + 3 * 512);
      c0 = __builtin_amdgcn_wmma_f32_16x16x32_f16(false, a0, false, b00, (short)0, c0, false, false);
      c1 = __builtin_amdgcn_wmma_f32_16x16x32_f16(false, a0, false, b01, (short)0, c1, false, false);
      c2 = __builtin_amdgcn_wmma_f32_16x16x32_f16(false, a0, false, b02, (short)0, c2, false, false);
      c3 = __builtin_amdgcn_wmma_f32_16x16x32_f16(false, a0, false, b03, (short)0, c3, false, false);
      if (kk + 2 < kcnt) {
        // load even set (kk+2) while computing odd set (kk+1)
        const _Float16* bT0 =
            bp + (((size_t)(kc + kk + 2) * Ntiles + ntBase) * 32 + lane) * 16;
        a0  = *(const v16h*)(AsLane + (kk + 2) * (32 * APITCH));
        b00 = *(const v16h*)(bT0 + 0 * 512);
        b01 = *(const v16h*)(bT0 + 1 * 512);
        b02 = *(const v16h*)(bT0 + 2 * 512);
        b03 = *(const v16h*)(bT0 + 3 * 512);
      }
      c0 = __builtin_amdgcn_wmma_f32_16x16x32_f16(false, a1, false, b10, (short)0, c0, false, false);
      c1 = __builtin_amdgcn_wmma_f32_16x16x32_f16(false, a1, false, b11, (short)0, c1, false, false);
      c2 = __builtin_amdgcn_wmma_f32_16x16x32_f16(false, a1, false, b12, (short)0, c2, false, false);
      c3 = __builtin_amdgcn_wmma_f32_16x16x32_f16(false, a1, false, b13, (short)0, c3, false, false);
    }
  }

  // D 16x16 f32 layout: lanes 0-15 -> rows 0..7, lanes 16-31 -> rows 8..15
  int mBase = mt * 16 + ((lane >> 4) << 3);
  int nr = lane & 15;
  v8f acc[4] = {c0, c1, c2, c3};
#pragma unroll
  for (int i = 0; i < 4; ++i) {
    int nOut = (ntBase + i) * 16 + nr;
#pragma unroll
    for (int v = 0; v < 8; ++v)
      out[(size_t)(mBase + v) * N + nOut] = acc[i][v];
  }
}

// ---------------------------------------------------------------------------
// SMALL GEMM (any Ntiles): one 16x16 tile per wave, scalar early-exit.
// Used only for the tiny levels (Ntiles = 16 or 4).
// ---------------------------------------------------------------------------
__global__ void __launch_bounds__(256)
wmma_gemm_f16_small(const unsigned short* __restrict__ ap_, const unsigned short* __restrict__ bp_,
                    float* __restrict__ out, int Ntiles, int Ktiles, int N) {
  const _Float16* __restrict__ ap = (const _Float16*)ap_;
  const _Float16* __restrict__ bp = (const _Float16*)bp_;
  int wave = blockIdx.x * 8 + __builtin_amdgcn_readfirstlane(threadIdx.x >> 5);
  int lane = threadIdx.x & 31;
  if (wave >= 16 * Ntiles) return;        // scalar (SGPR) condition
  int mt = wave / Ntiles, nt = wave % Ntiles;

  const _Float16* aBase = ap + (size_t)mt * Ktiles * 512 + (size_t)lane * 16;
  const _Float16* bBase = bp + ((size_t)nt * 32 + lane) * 16;
  v8f c = {};
  for (int kt = 0; kt < Ktiles; ++kt) {
    v16h a = *(const v16h*)(aBase + (size_t)kt * 512);
    v16h b = *(const v16h*)(bBase + (size_t)kt * Ntiles * 512);
    c = __builtin_amdgcn_wmma_f32_16x16x32_f16(false, a, false, b, (short)0, c, false, false);
  }
  int nOut  = nt * 16 + (lane & 15);
  int mBase = mt * 16 + ((lane >> 4) << 3);
#pragma unroll
  for (int v = 0; v < 8; ++v)
    out[(size_t)(mBase + v) * N + nOut] = c[v];
}

// ---------------------------------------------------------------------------
// Nearest-neighbor upsample of cat[i] into channels [256, 256+Csrc) of cat[i-1].
// ---------------------------------------------------------------------------
__global__ void upsample_concat(const float* __restrict__ src, float* __restrict__ dst,
                                int Bn, int Csrc, int Hin, int Win,
                                int Cdst, int Hout, int Wout) {
  long idx = (long)blockIdx.x * blockDim.x + threadIdx.x;
  long total = (long)Bn * Csrc * Hout * Wout;
  if (idx >= total) return;
  int xw = (int)(idx % Wout); long t = idx / Wout;
  int y = (int)(t % Hout); t /= Hout;
  int c = (int)(t % Csrc); int b = (int)(t / Csrc);
  int ys = (y * Hin) / Hout, xs = (xw * Win) / Wout;
  dst[(((size_t)b * Cdst + 256 + c) * Hout + y) * Wout + xw] =
      src[(((size_t)b * Csrc + c) * Hin + ys) * Win + xs];
}

// ---------------------------------------------------------------------------
// Host orchestration
// ---------------------------------------------------------------------------
static inline int cdiv(long a, int b) { return (int)((a + b - 1) / b); }

static void launch_gemm(const unsigned short* ap, const unsigned short* bp, float* dst,
                        int Ntiles, int Ktiles, int N, hipStream_t stream) {
  if (Ntiles % 32 == 0) {
    dim3 grid(Ntiles / 32, 16);
    wmma_gemm_f16_full<<<grid, 256, 0, stream>>>(ap, bp, dst, Ntiles, Ktiles, N);
  } else {
    wmma_gemm_f16_small<<<cdiv((long)16 * Ntiles, 8), 256, 0, stream>>>(
        ap, bp, dst, Ntiles, Ktiles, N);
  }
}

extern "C" void kernel_launch(void* const* d_in, const int* in_sizes, int n_in,
                              void* d_out, int out_size, void* d_ws, size_t ws_size,
                              hipStream_t stream) {
  (void)in_sizes; (void)n_in; (void)out_size; (void)ws_size;
  const int Bn = 2;
  const int INS[4]  = {256, 512, 1024, 2048};
  const int HS[4]   = {64, 32, 16, 8};
  const int FCIN[4] = {1024, 768, 512, 256};

  const float *x[4], *latw[4], *latow[4], *latob[4], *fpnw[4], *fpnow[4], *fpnob[4];
  for (int i = 0; i < 4; ++i) {
    x[i]     = (const float*)d_in[i];
    latw[i]  = (const float*)d_in[4 + i];
    latow[i] = (const float*)d_in[8 + i];
    latob[i] = (const float*)d_in[12 + i];
    fpnw[i]  = (const float*)d_in[16 + i];
    fpnow[i] = (const float*)d_in[20 + i];
    fpnob[i] = (const float*)d_in[24 + i];
  }
  float* out = (float*)d_out;

  // ---- workspace carve (256B aligned) ----
  char* wsp = (char*)d_ws;
  auto carve = [&](size_t bytes) -> char* {
    char* p = wsp;
    wsp += (bytes + 255) & ~(size_t)255;
    return p;
  };
  float* cat[4];
  for (int i = 0; i < 4; ++i)
    cat[i] = (float*)carve((size_t)Bn * FCIN[i] * HS[i] * HS[i] * sizeof(float));
  float* off = (float*)carve((size_t)Bn * 18 * 64 * 64 * sizeof(float));
  unsigned short *apLat[4], *apFpn[4];
  for (int i = 0; i < 4; ++i)
    apLat[i] = (unsigned short*)carve((size_t)256 * INS[i] * 9 * sizeof(unsigned short));
  for (int i = 0; i < 4; ++i)
    apFpn[i] = (unsigned short*)carve((size_t)256 * FCIN[i] * 9 * sizeof(unsigned short));
  unsigned short* bpack = (unsigned short*)carve((size_t)9216 * 4096 * sizeof(unsigned short));

  const int TB = 256;

  // ---- pack all weights to WMMA A-fragment f16 ----
  for (int i = 0; i < 4; ++i) {
    long nl = (long)256 * INS[i] * 9;
    pack_A_f16<<<cdiv(nl, TB), TB, 0, stream>>>(latw[i], apLat[i], 256, INS[i] * 9);
    long nf = (long)256 * FCIN[i] * 9;
    pack_A_f16<<<cdiv(nf, TB), TB, 0, stream>>>(fpnw[i], apFpn[i], 256, FCIN[i] * 9);
  }

  // ---- lateral DCNs -> cat[i] channels [0,256) ----
  for (int i = 0; i < 4; ++i) {
    int H = HS[i], W = HS[i], HW = H * W, Cin = INS[i];
    int Kdim = Cin * 9, Ktiles = Kdim >> 5, Ntiles = HW >> 4;
    offset_conv<<<cdiv((long)Bn * 18 * HW, TB), TB, 0, stream>>>(
        x[i], latow[i], latob[i], off, Bn, Cin, H, W);
    for (int b = 0; b < Bn; ++b) {
      sample_pack_B<<<cdiv(((long)Kdim * HW) >> 4, TB), TB, 0, stream>>>(
          x[i] + (size_t)b * Cin * HW, off + (size_t)b * 18 * HW, bpack, Cin, H, W);
      float* dst = cat[i] + (size_t)b * FCIN[i] * HW;
      launch_gemm(apLat[i], bpack, dst, Ntiles, Ktiles, HW, stream);
    }
  }

  // ---- top-down: cat[i] -> nearest-up -> cat[i-1] ch 256.. ----
  for (int i = 3; i >= 1; --i) {
    int Csrc = FCIN[i], Hin = HS[i], Hout = HS[i - 1];
    upsample_concat<<<cdiv((long)Bn * Csrc * Hout * Hout, TB), TB, 0, stream>>>(
        cat[i], cat[i - 1], Bn, Csrc, Hin, Hin, FCIN[i - 1], Hout, Hout);
  }

  // ---- FPN DCNs -> d_out ----
  size_t outOff = 0;
  for (int i = 0; i < 4; ++i) {
    int H = HS[i], W = HS[i], HW = H * W, Cin = FCIN[i];
    int Kdim = Cin * 9, Ktiles = Kdim >> 5, Ntiles = HW >> 4;
    offset_conv<<<cdiv((long)Bn * 18 * HW, TB), TB, 0, stream>>>(
        cat[i], fpnow[i], fpnob[i], off, Bn, Cin, H, W);
    for (int b = 0; b < Bn; ++b) {
      sample_pack_B<<<cdiv(((long)Kdim * HW) >> 4, TB), TB, 0, stream>>>(
          cat[i] + (size_t)b * Cin * HW, off + (size_t)b * 18 * HW, bpack, Cin, H, W);
      float* dst = out + outOff + (size_t)b * 256 * HW;
      launch_gemm(apFpn[i], bpack, dst, Ntiles, Ktiles, HW, stream);
    }
    outOff += (size_t)Bn * 256 * HW;
  }
}
